// Conv1d_NN_18657337934497
// MI455X (gfx1250) — matrix operand order repaired and checked by reading, not compiled, not run on hardware
//
#include <hip/hip_runtime.h>
#include <float.h>

// Problem constants (match the JAX reference).
constexpr int B_    = 16;
constexpr int C_    = 64;
constexpr int N_    = 2048;
constexpr int KNN_  = 3;     // neighbors == conv kernel width == conv stride
constexpr int COUT_ = 64;

typedef __attribute__((ext_vector_type(8)))  float  v8f;
typedef __attribute__((ext_vector_type(8)))  __bf16 v8bf;
typedef __attribute__((ext_vector_type(16))) __bf16 v16bf;

#define DEV_INLINE __device__ __forceinline__

// ---------------------------------------------------------------------------
// Kernel A: transpose x [B,C,N] -> point-major xt [B,N,C] in three forms:
//   xt_f32 (exact, for the conv gather), xt_hi = bf16(x), xt_lo = bf16(x-hi)
// (split-bf16 operands for the WMMA Gram matrix), plus per-point sq-norms.
// ---------------------------------------------------------------------------
__global__ __launch_bounds__(256) void xpose_norm_kernel(
    const float* __restrict__ x, float* __restrict__ xt,
    __bf16* __restrict__ xt_hi, __bf16* __restrict__ xt_lo,
    float* __restrict__ norms) {
  int tid = blockIdx.x * blockDim.x + threadIdx.x;
  if (tid >= B_ * N_) return;
  int b = tid / N_;
  int n = tid - b * N_;
  const float* src = x + (size_t)b * C_ * N_ + n;   // coalesced over n
  float*  df = xt    + (size_t)tid * C_;
  __bf16* dh = xt_hi + (size_t)tid * C_;
  __bf16* dl = xt_lo + (size_t)tid * C_;
  float s = 0.f;
#pragma unroll
  for (int c = 0; c < C_; ++c) {
    float v = src[(size_t)c * N_];
    df[c] = v;
    __bf16 h = (__bf16)v;            // RNE truncation to bf16
    __bf16 l = (__bf16)(v - (float)h);
    dh[c] = h;
    dl[c] = l;
    s = fmaf(v, v, s);
  }
  norms[tid] = s;
}

// Selects that must stay v_cndmask: __builtin_unpredictable suppresses
// clang's select->branch (EXEC-diamond) conversion seen at -O3.
DEV_INLINE float fsel(bool c, float a, float b) {
  return __builtin_unpredictable(c) ? a : b;
}
DEV_INLINE int isel(bool c, int a, int b) {
  return __builtin_unpredictable(c) ? a : b;
}

// Branchless insertion into a sorted (ascending) 3-element min-list.
DEV_INLINE void top3_insert(float d, int mi,
                            float& d0, float& d1, float& d2,
                            int& i0, int& i1, int& i2) {
  bool lt2 = d < d2, lt1 = d < d1, lt0 = d < d0;
  float nd2 = fsel(lt1, d1, fminf(d, d2));
  int   ni2 = isel(lt1, i1, isel(lt2, mi, i2));
  float nd1 = fsel(lt0, d0, fminf(d, d1));
  int   ni1 = isel(lt0, i0, isel(lt1, mi, i1));
  float nd0 = fminf(d, d0);
  int   ni0 = isel(lt0, mi, i0);
  d0 = nd0; d1 = nd1; d2 = nd2;
  i0 = ni0; i1 = ni1; i2 = ni2;
}

// A-fragment loader for V_WMMA_F32_16X16X32_BF16 (16-bit A 16x32 layout):
// lane half h holds K = [16h, 16h+16) of this 32-K chunk (contiguous, matches B).
DEV_INLINE v16bf load_a_frag(const __bf16* chunk_base, int h) {
  v8bf a = *reinterpret_cast<const v8bf*>(chunk_base + 16 * h);
  v8bf b = *reinterpret_cast<const v8bf*>(chunk_base + 16 * h + 8);
  return __builtin_shufflevector(a, b, 0, 1, 2, 3, 4, 5, 6, 7,
                                       8, 9, 10, 11, 12, 13, 14, 15);
}

// ---------------------------------------------------------------------------
// Kernel B: fused pairwise-distance + streaming top-3 on the bf16 matrix pipe.
// One wave owns 32 query points (two 16-wide B-fragment sets), so each
// streamed 16-candidate tile (8x b128 loads, shared) feeds 12
// V_WMMA_F32_16X16X32_BF16 issues. dot_f32 is emulated as
//     hi_m*hi_q + hi_m*lo_q + lo_m*hi_q        (split-bf16, ~2^-16 rel. err)
// Ordering key per query is nm - 2*dot (query norm is a monotonic shift and
// is dropped entirely).
//
// Fragment layout (wave32, bf16 16x16x32), h = lane>>4:
//   A (candidates 16x32): lane row M = lane&15, K = 16h..16h+15 (contiguous)
//   B (queries    32x16): lane col   = lane&15, K = 16h..16h+15 (contiguous)
//   D (16x16 f32):        lane holds rows 8h+r (r=0..7) of its query column
// ---------------------------------------------------------------------------
__global__ __launch_bounds__(32) void knn_wmma_kernel(
    const __bf16* __restrict__ xt_hi, const __bf16* __restrict__ xt_lo,
    const float* __restrict__ norms, int* __restrict__ knn_idx) {
  const int lane = threadIdx.x;
  const int h    = lane >> 4;
  const int l15  = lane & 15;
  const int b    = blockIdx.y;

  const __bf16* hi_b    = xt_hi + (size_t)b * N_ * C_;
  const __bf16* lo_b    = xt_lo + (size_t)b * N_ * C_;
  const float*  norms_b = norms + (size_t)b * N_;

  // Register-resident B fragments: 2 query tiles x 2 K-chunks (C=64).
  v16bf bq_hi[2][2], bq_lo[2][2];
#pragma unroll
  for (int qt = 0; qt < 2; ++qt) {
    const int q = blockIdx.x * 32 + qt * 16 + l15;
#pragma unroll
    for (int ch = 0; ch < 2; ++ch) {
      size_t off = (size_t)q * C_ + 32 * ch + 16 * h;
      bq_hi[qt][ch] = *reinterpret_cast<const v16bf*>(hi_b + off);
      bq_lo[qt][ch] = *reinterpret_cast<const v16bf*>(lo_b + off);
    }
  }

  float d0[2] = {FLT_MAX, FLT_MAX}, d1[2] = {FLT_MAX, FLT_MAX},
        d2[2] = {FLT_MAX, FLT_MAX};
  int   i0[2] = {0, 0}, i1[2] = {0, 0}, i2[2] = {0, 0};

  for (int m0 = 0; m0 < N_; m0 += 16) {
    const __bf16* arow_hi = hi_b + (size_t)(m0 + l15) * C_;
    const __bf16* arow_lo = lo_b + (size_t)(m0 + l15) * C_;
    if (m0 + 16 < N_) {  // gfx1250 global_prefetch of next candidate tile
      __builtin_prefetch(arow_hi + 16 * C_, 0, 3);
      __builtin_prefetch(arow_lo + 16 * C_, 0, 3);
    }

    v8f acc[2] = {v8f{}, v8f{}};
#pragma unroll
    for (int ch = 0; ch < 2; ++ch) {
      v16bf a_hi = load_a_frag(arow_hi + 32 * ch, h);
      v16bf a_lo = load_a_frag(arow_lo + 32 * ch, h);
#pragma unroll
      for (int qt = 0; qt < 2; ++qt) {
        acc[qt] = __builtin_amdgcn_wmma_f32_16x16x32_bf16(
            false, a_hi, false, bq_hi[qt][ch], (short)0, acc[qt], false, false);
        acc[qt] = __builtin_amdgcn_wmma_f32_16x16x32_bf16(
            false, a_hi, false, bq_lo[qt][ch], (short)0, acc[qt], false, false);
        acc[qt] = __builtin_amdgcn_wmma_f32_16x16x32_bf16(
            false, a_lo, false, bq_hi[qt][ch], (short)0, acc[qt], false, false);
      }
    }

    // Candidate norms for the 8 rows this lane holds (shared by both tiles).
    const int mbase = m0 + h * 8;
    const float4* np4 = reinterpret_cast<const float4*>(norms_b + mbase);
    float4 na0 = np4[0], na1 = np4[1];
    float nm[8] = {na0.x, na0.y, na0.z, na0.w, na1.x, na1.y, na1.z, na1.w};
#pragma unroll
    for (int qt = 0; qt < 2; ++qt)
#pragma unroll
      for (int r = 0; r < 8; ++r) {
        float d = fmaf(-2.0f, acc[qt][r], nm[r]);   // ordering key: nm - 2*dot
        top3_insert(d, mbase + r, d0[qt], d1[qt], d2[qt],
                    i0[qt], i1[qt], i2[qt]);
      }
  }

  // Merge the two half-wave candidate lists (rows 0-7 vs rows 8-15) per tile.
#pragma unroll
  for (int qt = 0; qt < 2; ++qt) {
    float e0 = __shfl_xor(d0[qt], 16, 32); int j0 = __shfl_xor(i0[qt], 16, 32);
    float e1 = __shfl_xor(d1[qt], 16, 32); int j1 = __shfl_xor(i1[qt], 16, 32);
    float e2 = __shfl_xor(d2[qt], 16, 32); int j2 = __shfl_xor(i2[qt], 16, 32);
    top3_insert(e0, j0, d0[qt], d1[qt], d2[qt], i0[qt], i1[qt], i2[qt]);
    top3_insert(e1, j1, d0[qt], d1[qt], d2[qt], i0[qt], i1[qt], i2[qt]);
    top3_insert(e2, j2, d0[qt], d1[qt], d2[qt], i0[qt], i1[qt], i2[qt]);
    if (lane < 16) {
      const int q = blockIdx.x * 32 + qt * 16 + l15;
      int* o = knn_idx + ((size_t)b * N_ + q) * KNN_;
      o[0] = i0[qt]; o[1] = i1[qt]; o[2] = i2[qt];
    }
  }
}

// ---------------------------------------------------------------------------
// Kernel C: gather the 3 neighbor feature vectors (192 f32) into LDS and run
// the [64 x 192] conv as a dot per (co, n) + bias + ReLU, in full f32. Tiny
// FLOP count relative to Kernel B; LDS reads broadcast across the 64 co
// threads of a point; W (48 KB) stays hot in cache.
// ---------------------------------------------------------------------------
__global__ __launch_bounds__(256) void gather_conv_kernel(
    const float* __restrict__ xt, const int* __restrict__ knn_idx,
    const float* __restrict__ W, const float* __restrict__ bias,
    float* __restrict__ out) {
  __shared__ float prime[4][C_ * KNN_];   // 4 points x 192 floats
  const int t  = threadIdx.x;
  const int co = t & 63;
  const int p  = t >> 6;                  // point within block tile (0..3)
  const int b  = blockIdx.y;
  const int n  = blockIdx.x * 4 + p;

  const int* id = knn_idx + ((size_t)b * N_ + n) * KNN_;
  int nb0 = id[0], nb1 = id[1], nb2 = id[2];
#pragma unroll
  for (int j = co; j < C_ * KNN_; j += 64) {
    int c = j / 3;
    int k = j - c * 3;
    int nb = (k == 0) ? nb0 : ((k == 1) ? nb1 : nb2);
    prime[p][j] = xt[((size_t)b * N_ + nb) * C_ + c];  // prime[c*3+k] = x[:, idx[k]]
  }
  __syncthreads();

  float acc = bias[co];
  const float* Wr = W + (size_t)co * (C_ * KNN_);      // W[co][c][k], j = c*3+k
#pragma unroll 8
  for (int j = 0; j < C_ * KNN_; ++j)
    acc = fmaf(Wr[j], prime[p][j], acc);

  out[((size_t)b * COUT_ + co) * N_ + n] = fmaxf(acc, 0.f);
}

// ---------------------------------------------------------------------------
extern "C" void kernel_launch(void* const* d_in, const int* in_sizes, int n_in,
                              void* d_out, int out_size, void* d_ws, size_t ws_size,
                              hipStream_t stream) {
  (void)in_sizes; (void)n_in; (void)out_size; (void)ws_size;
  const float* x    = (const float*)d_in[0];  // [B, C, N]
  const float* W    = (const float*)d_in[1];  // [COUT, C, K]
  const float* bias = (const float*)d_in[2];  // [COUT]
  float* out = (float*)d_out;                 // [B, COUT, N]

  // Workspace: xt f32 (8MB) | xt_hi bf16 (4MB) | xt_lo bf16 (4MB)
  //            | norms f32 (128KB) | knn_idx i32 (384KB)
  float*  xt      = (float*)d_ws;
  __bf16* xt_hi   = (__bf16*)(xt + (size_t)B_ * N_ * C_);
  __bf16* xt_lo   = xt_hi + (size_t)B_ * N_ * C_;
  float*  norms   = (float*)(xt_lo + (size_t)B_ * N_ * C_);
  int*    knn_idx = (int*)(norms + (size_t)B_ * N_);

  xpose_norm_kernel<<<(B_ * N_ + 255) / 256, 256, 0, stream>>>(
      x, xt, xt_hi, xt_lo, norms);
  knn_wmma_kernel<<<dim3(N_ / 32, B_), 32, 0, stream>>>(
      xt_hi, xt_lo, norms, knn_idx);
  gather_conv_kernel<<<dim3(N_ / 4, B_), 256, 0, stream>>>(
      xt, knn_idx, W, bias, out);
}
